// EmbeddedAttention_21199958573740
// MI455X (gfx1250) — matrix operand order repaired and checked by reading, not compiled, not run on hardware
//
#include <hip/hip_runtime.h>
#include <hip/hip_bf16.h>

// EmbeddedAttention, MI455X (gfx1250), wave32, WMMA bf16 path.
// out[b,n] = (Q_n @ (K_n @ value[b,n])) @ Wv + bv   (softmax row-sums == 1)
// B=32 N=12 S=512 D=E=256 A=96

#define B_ 32
#define N_ 12
#define S_ 512
#define D_ 256
#define A_ 96

typedef __attribute__((ext_vector_type(16))) __bf16 v16bf;
typedef __attribute__((ext_vector_type(8)))  float  v8f;
typedef __attribute__((ext_vector_type(8)))  unsigned int v8u;
typedef __attribute__((ext_vector_type(4)))  int v4i;

static_assert(sizeof(v16bf) == 32, "v16bf must be 32 bytes");
static_assert(sizeof(v8f) == 32, "v8f must be 32 bytes");

#if defined(__has_builtin)
#if __has_builtin(__builtin_amdgcn_global_load_async_to_lds_b128)
#define HAVE_ASYNC_LDS 1
#endif
#endif

// float -> bf16 (native convert where available)
__device__ __forceinline__ unsigned short f2bf(float f) {
    __hip_bfloat16 h = __float2bfloat16(f);
    return __builtin_bit_cast(unsigned short, h);
}

// Build a v16bf fragment from two 16-byte chunks (elements 0..7 = lo, 8..15 = hi).
__device__ __forceinline__ v16bf make_frag(uint4 lo, uint4 hi) {
    v8u t;
    t[0] = lo.x; t[1] = lo.y; t[2] = lo.z; t[3] = lo.w;
    t[4] = hi.x; t[5] = hi.y; t[6] = hi.z; t[7] = hi.w;
    return __builtin_bit_cast(v16bf, t);
}

__device__ __forceinline__ v8f wmma_bf16(v16bf a, v16bf b, v8f c) {
    return __builtin_amdgcn_wmma_f32_16x16x32_bf16(false, a, false, b, (short)0, c, false, false);
}

// ---------------------------------------------------------------------------
// Kernel 1: qk[n,s,d] = emb[n,s,:] @ Wqk[:,d] + bqk[d]   (fp32, tiny)
// ---------------------------------------------------------------------------
__global__ void k_qk(const float* __restrict__ emb, const float* __restrict__ Wqk,
                     const float* __restrict__ bqk, float* __restrict__ qk) {
    __shared__ float es[A_];
    int row = blockIdx.x;          // n*S_ + s
    int t = threadIdx.x;           // d
    if (t < A_) es[t] = emb[row * A_ + t];
    __syncthreads();
    float acc = bqk[t];
    #pragma unroll 4
    for (int a = 0; a < A_; ++a) acc += es[a] * Wqk[a * D_ + t];
    qk[row * D_ + t] = acc;
}

// ---------------------------------------------------------------------------
// Kernel 2: Qbf[n,s,d] = bf16(softmax_d(qk[n,s,:]))
// ---------------------------------------------------------------------------
__global__ void k_softmax_q(const float* __restrict__ qk, unsigned short* __restrict__ Qbf) {
    __shared__ float red[256];
    int row = blockIdx.x;
    int t = threadIdx.x;
    float x = qk[row * D_ + t];
    red[t] = x; __syncthreads();
    for (int off = 128; off > 0; off >>= 1) {
        if (t < off) red[t] = fmaxf(red[t], red[t + off]);
        __syncthreads();
    }
    float m = red[0]; __syncthreads();
    float e = __expf(x - m);
    red[t] = e; __syncthreads();
    for (int off = 128; off > 0; off >>= 1) {
        if (t < off) red[t] += red[t + off];
        __syncthreads();
    }
    float inv = 1.0f / red[0];
    Qbf[row * D_ + t] = f2bf(e * inv);
}

// ---------------------------------------------------------------------------
// Kernel 3: Ktbf[n,d,s] = bf16(softmax_s(qk[n,:,d]))   (stored transposed)
// ---------------------------------------------------------------------------
__global__ void k_softmax_k(const float* __restrict__ qk, unsigned short* __restrict__ Ktbf) {
    __shared__ float red[256];
    int n = blockIdx.x / D_;
    int d = blockIdx.x % D_;
    int t = threadIdx.x;
    const float* base = qk + (size_t)(n * S_) * D_ + d;
    float x0 = base[(size_t)t * D_];
    float x1 = base[(size_t)(t + 256) * D_];
    red[t] = fmaxf(x0, x1); __syncthreads();
    for (int off = 128; off > 0; off >>= 1) {
        if (t < off) red[t] = fmaxf(red[t], red[t + off]);
        __syncthreads();
    }
    float m = red[0]; __syncthreads();
    float e0 = __expf(x0 - m), e1 = __expf(x1 - m);
    red[t] = e0 + e1; __syncthreads();
    for (int off = 128; off > 0; off >>= 1) {
        if (t < off) red[t] += red[t + off];
        __syncthreads();
    }
    float inv = 1.0f / red[0];
    unsigned short* o = Ktbf + (size_t)(n * D_ + d) * S_;
    o[t] = f2bf(e0 * inv);
    o[t + 256] = f2bf(e1 * inv);
}

// ---------------------------------------------------------------------------
// Kernel 4: Wvt[e,d] = bf16(Wv[d,e])   (transposed for contiguous B-fragments)
// ---------------------------------------------------------------------------
__global__ void k_wvt(const float* __restrict__ Wv, unsigned short* __restrict__ Wvt) {
    int e = blockIdx.x, t = threadIdx.x;
    Wvt[(size_t)e * D_ + t] = f2bf(Wv[(size_t)t * D_ + e]);
}

// ---------------------------------------------------------------------------
// Kernel 5 (heavy): per (b,n): T1 = K[d,s] @ bf16(value[s,e]), written
// transposed as T1t[b,n][e][d] bf16.  16 waves; wave w owns rows 16w..16w+15.
// value tile [32 s][256 e] double-buffered:  issue(global loads) -> WMMA block
// (hides HBM latency) -> commit(convert+LDS store).  B-frags depth-2 pipelined.
// ---------------------------------------------------------------------------
#define VST 40   // LDS row stride (bf16 elems) for transposed V tile (32 + 8 pad)

__global__ void __launch_bounds__(512, 1)
k_kv(const float* __restrict__ value, const unsigned short* __restrict__ Ktbf,
     unsigned short* __restrict__ T1t) {
    extern __shared__ unsigned short lds[];   // 2 buffers * 256*VST bf16 = 40 KB
    int bn = blockIdx.x;
    int n = bn % N_;
    const float* V = value + (size_t)bn * S_ * D_;
    const unsigned short* Kt = Ktbf + (size_t)n * D_ * S_;   // [d][s], stride S_

    int tid  = threadIdx.x;
    int wave = tid >> 5, lane = tid & 31;
    int l15 = lane & 15, hi = lane >> 4;
    int m0 = wave * 16;                        // T1 row strip (d dimension)

    int dsr = tid >> 4;                        // 0..31  (s within tile)
    int e0  = (tid & 15) * 16;                 // 0..240 (e chunk base)

    v8f acc[16];
    #pragma unroll
    for (int i = 0; i < 16; ++i) acc[i] = {};

    float4 r0, r1, r2, r3;                     // staged tile chunk in registers
    auto issue = [&](int s0) {                 // global loads only (no wait yet)
        const float4* src = (const float4*)(V + (size_t)(s0 + dsr) * D_ + e0);
        r0 = src[0]; r1 = src[1]; r2 = src[2]; r3 = src[3];
    };
    auto commit = [&](int buf) {               // convert + transposed LDS store
        unsigned short* dst = lds + buf * (256 * VST) + dsr;
        dst[(e0 +  0) * VST] = f2bf(r0.x); dst[(e0 +  1) * VST] = f2bf(r0.y);
        dst[(e0 +  2) * VST] = f2bf(r0.z); dst[(e0 +  3) * VST] = f2bf(r0.w);
        dst[(e0 +  4) * VST] = f2bf(r1.x); dst[(e0 +  5) * VST] = f2bf(r1.y);
        dst[(e0 +  6) * VST] = f2bf(r1.z); dst[(e0 +  7) * VST] = f2bf(r1.w);
        dst[(e0 +  8) * VST] = f2bf(r2.x); dst[(e0 +  9) * VST] = f2bf(r2.y);
        dst[(e0 + 10) * VST] = f2bf(r2.z); dst[(e0 + 11) * VST] = f2bf(r2.w);
        dst[(e0 + 12) * VST] = f2bf(r3.x); dst[(e0 + 13) * VST] = f2bf(r3.y);
        dst[(e0 + 14) * VST] = f2bf(r3.z); dst[(e0 + 15) * VST] = f2bf(r3.w);
    };
    auto loadA = [&](int s0) -> v16bf {
        const unsigned short* ap = Kt + (size_t)(m0 + l15) * S_ + s0 + hi * 16;
        return make_frag(*(const uint4*)ap, *(const uint4*)(ap + 8));
    };
    auto loadB = [&](const unsigned short* vb, int nt) -> v16bf {
        const unsigned short* bp = vb + (nt * 16 + l15) * VST;
        return make_frag(*(const uint4*)bp, *(const uint4*)(bp + 8));
    };

    issue(0); commit(0);
    v16bf a = loadA(0);
    __syncthreads();

    for (int ks = 0; ks < 16; ++ks) {          // S_/32 K-steps
        if (ks + 1 < 16) issue((ks + 1) * 32);             // HBM fetch in flight
        if (ks + 2 < 16)
            __builtin_prefetch(V + (size_t)((ks + 2) * 32 + dsr) * D_ + e0, 0, 0);
        v16bf an = a;
        if (ks + 1 < 16) an = loadA((ks + 1) * 32);        // next A in flight

        const unsigned short* vb = lds + (ks & 1) * (256 * VST) + hi * 16;
        v16bf b0 = loadB(vb, 0);
        v16bf b1 = loadB(vb, 1);
        #pragma unroll
        for (int nt = 0; nt < 16; ++nt) {
            v16bf b2 = b1;
            if (nt + 2 < 16) b2 = loadB(vb, nt + 2);       // depth-2 pipeline
            acc[nt] = wmma_bf16(a, b0, acc[nt]);
            b0 = b1; b1 = b2;
        }
        a = an;
        if (ks + 1 < 16) commit((ks + 1) & 1);             // waits HBM here
        __syncthreads();
    }

    // Store transposed: T1t[e][d]; accumulator VGPR r = consecutive d -> packed b128
    unsigned short* T1 = T1t + (size_t)bn * D_ * D_;
    #pragma unroll
    for (int nt = 0; nt < 16; ++nt) {
        int e  = nt * 16 + l15;
        int d0 = m0 + hi * 8;
        uint4 pk;
        pk.x = (unsigned)f2bf(acc[nt][0]) | ((unsigned)f2bf(acc[nt][1]) << 16);
        pk.y = (unsigned)f2bf(acc[nt][2]) | ((unsigned)f2bf(acc[nt][3]) << 16);
        pk.z = (unsigned)f2bf(acc[nt][4]) | ((unsigned)f2bf(acc[nt][5]) << 16);
        pk.w = (unsigned)f2bf(acc[nt][6]) | ((unsigned)f2bf(acc[nt][7]) << 16);
        *(uint4*)(T1 + (size_t)e * D_ + d0) = pk;
    }
}

// ---------------------------------------------------------------------------
// Kernel 6 (heavy): per (b,n): out = (Q @ T1) @ Wv(^T as Wvt) + bv.
// T1t staged to LDS via async global->LDS copy (ASYNCcnt) when available.
// ---------------------------------------------------------------------------
#define T1ST 272   // 256 + 16 pad (bf16 elems)
#define SCST 264   // 256 + 8 pad  (bf16 elems)

__global__ void __launch_bounds__(512, 1)
k_out(const unsigned short* __restrict__ Qbf, const unsigned short* __restrict__ T1t,
      const unsigned short* __restrict__ Wvt, const float* __restrict__ bv,
      float* __restrict__ out) {
    extern __shared__ unsigned short lds[];
    unsigned short* T1s = lds;                       // 256 x T1ST
    unsigned short* scr = lds + 256 * T1ST;          // 16 waves x 16 x SCST

    int bn = blockIdx.x;
    int n = bn % N_;
    const unsigned short* Q  = Qbf + (size_t)n * S_ * D_;
    const unsigned short* T1 = T1t + (size_t)bn * D_ * D_;
    float* O = out + (size_t)bn * S_ * D_;

    int tid = threadIdx.x, wave = tid >> 5, lane = tid & 31;
    int l15 = lane & 15, hi = lane >> 4;

    // ---- Stage T1t (256x256 bf16 = 128 KB) into LDS with padded stride ----
#if defined(HAVE_ASYNC_LDS)
    for (int c = tid; c < 8192; c += 512) {          // 8192 16-byte chunks
        int row = c >> 5;
        int col = (c & 31) * 8;
        // builtin signature (from diagnostics): (v4i* src, v4i* dst, imm offset, imm cpol)
        __builtin_amdgcn_global_load_async_to_lds_b128(
            (v4i*)(T1 + (size_t)row * D_ + col),
            (v4i*)(T1s + row * T1ST + col),
            0, 0);
    }
    asm volatile("s_wait_asynccnt 0x0" ::: "memory");
#else
    for (int c = tid; c < 8192; c += 512) {
        int row = c >> 5;
        int col = (c & 31) * 8;
        uint4 v = *(const uint4*)(T1 + (size_t)row * D_ + col);
        *(uint4*)(T1s + row * T1ST + col) = v;
    }
#endif
    __syncthreads();

    unsigned short* myscr = scr + wave * (16 * SCST);

    auto loadQ = [&](int m0, int k0) -> v16bf {
        const unsigned short* ap = Q + (size_t)(m0 + l15) * D_ + k0 + hi * 16;
        return make_frag(*(const uint4*)ap, *(const uint4*)(ap + 8));
    };
    auto loadT = [&](int k0, int nt) -> v16bf {
        const unsigned short* bp = T1s + (nt * 16 + l15) * T1ST + k0 + hi * 16;
        return make_frag(*(const uint4*)bp, *(const uint4*)(bp + 8));
    };
    auto loadW = [&](int k0, int nt) -> v16bf {
        const unsigned short* bp = Wvt + (size_t)(nt * 16 + l15) * D_ + k0 + hi * 16;
        return make_frag(*(const uint4*)bp, *(const uint4*)(bp + 8));
    };
    auto loadS = [&](int k0) -> v16bf {
        const unsigned short* ap = myscr + l15 * SCST + k0 + hi * 16;
        return make_frag(*(const uint4*)ap, *(const uint4*)(ap + 8));
    };

    for (int iter = 0; iter < 2; ++iter) {
        int m0 = (wave + iter * 16) * 16;            // output row strip (s dim)

        // ---- GEMM 1: T2_strip[16,256] = Q_strip @ T1  (K = d = 256) ----
        v8f acc[16];
        #pragma unroll
        for (int i = 0; i < 16; ++i) acc[i] = {};
        v16bf a = loadQ(m0, 0);
        for (int k0 = 0; k0 < D_; k0 += 32) {
            v16bf an = a;
            if (k0 + 32 < D_) an = loadQ(m0, k0 + 32);     // next A in flight
            v16bf b0 = loadT(k0, 0);
            v16bf b1 = loadT(k0, 1);
            #pragma unroll
            for (int nt = 0; nt < 16; ++nt) {
                v16bf b2 = b1;
                if (nt + 2 < 16) b2 = loadT(k0, nt + 2);   // depth-2 pipeline
                acc[nt] = wmma_bf16(a, b0, acc[nt]);
                b0 = b1; b1 = b2;
            }
            a = an;
        }
        // C-layout -> row-major bf16 strip in per-wave scratch (A-layout source)
        #pragma unroll
        for (int nt = 0; nt < 16; ++nt) {
            #pragma unroll
            for (int r = 0; r < 8; ++r) {
                myscr[(r + hi * 8) * SCST + nt * 16 + l15] = f2bf(acc[nt][r]);
            }
        }
        __syncthreads();

        // ---- GEMM 2: out_strip = T2_strip @ Wv (B-frags from L2-hot Wvt) + bv ----
        #pragma unroll
        for (int i = 0; i < 16; ++i) acc[i] = {};
        for (int k0 = 0; k0 < D_; k0 += 32) {
            v16bf a2 = loadS(k0);
            v16bf w0 = loadW(k0, 0);
            v16bf w1 = loadW(k0, 1);
            v16bf w2 = loadW(k0, 2);
            #pragma unroll
            for (int nt = 0; nt < 16; ++nt) {
                v16bf w3 = w2;
                if (nt + 3 < 16) w3 = loadW(k0, nt + 3);   // depth-3 pipeline
                acc[nt] = wmma_bf16(a2, w0, acc[nt]);
                w0 = w1; w1 = w2; w2 = w3;
            }
        }
        #pragma unroll
        for (int nt = 0; nt < 16; ++nt) {
            int e = nt * 16 + l15;
            float bias = bv[e];
            #pragma unroll
            for (int r = 0; r < 8; ++r) {
                O[(size_t)(m0 + r + hi * 8) * D_ + e] = acc[nt][r] + bias;
            }
        }
        __syncthreads();
    }
}

// ---------------------------------------------------------------------------
// Host launch
// ---------------------------------------------------------------------------
extern "C" void kernel_launch(void* const* d_in, const int* in_sizes, int n_in,
                              void* d_out, int out_size, void* d_ws, size_t ws_size,
                              hipStream_t stream) {
    (void)in_sizes; (void)n_in; (void)out_size; (void)ws_size;

    const float* value = (const float*)d_in[0];
    const float* emb   = (const float*)d_in[1];
    const float* Wqk   = (const float*)d_in[2];
    const float* bqk   = (const float*)d_in[3];
    const float* Wv    = (const float*)d_in[4];
    const float* bv    = (const float*)d_in[5];
    float* out = (float*)d_out;

    // Workspace layout (bytes). Requires ~60.2 MB.
    char* ws = (char*)d_ws;
    float*          qk   = (float*)(ws);                         //  6,291,456 B
    unsigned short* Qbf  = (unsigned short*)(ws + 6291456);      //  3,145,728 B
    unsigned short* Ktbf = (unsigned short*)(ws + 9437184);      //  3,145,728 B
    unsigned short* Wvt  = (unsigned short*)(ws + 12582912);     //    131,072 B
    unsigned short* T1t  = (unsigned short*)(ws + 12713984);     // 50,331,648 B

    k_qk<<<N_ * S_, 256, 0, stream>>>(emb, Wqk, bqk, qk);
    k_softmax_q<<<N_ * S_, 256, 0, stream>>>(qk, Qbf);
    k_softmax_k<<<N_ * D_, 256, 0, stream>>>(qk, Ktbf);
    k_wvt<<<D_, 256, 0, stream>>>(Wv, Wvt);

    size_t lds5 = 2u * 256u * VST * sizeof(unsigned short);                    // 40 KB
    k_kv<<<B_ * N_, 512, lds5, stream>>>(value, Ktbf, T1t);

    size_t lds6 = (256u * T1ST + 16u * 16u * SCST) * sizeof(unsigned short);   // ~268 KB
    k_out<<<B_ * N_, 512, lds6, stream>>>(Qbf, T1t, Wvt, bv, out);
}
